// MultiHeadAttention_70643622085339
// MI455X (gfx1250) — compile-verified, matrix operand order checked by reading, not run on hardware
//
#include <hip/hip_runtime.h>
#include <hip/hip_bf16.h>

typedef __attribute__((ext_vector_type(16))) __bf16 bf16x16;
typedef __attribute__((ext_vector_type(8)))  __bf16 bf16x8;
typedef __attribute__((ext_vector_type(8)))  float  f32x8;

#define Bv   128
#define Nv   64
#define Dv   768
#define Hv   32
#define DKv  24
#define DKP  32      // padded DK
#define TEv  32
#define TDv  40
#define Lv   4
#define QKVN 2304    // 3*H*DK
#define LSTR 40      // LDS tile row stride (elements): bank-conflict-free for b128

__device__ __forceinline__ f32x8 wmma_bf16(bf16x16 a, bf16x16 b, f32x8 c) {
    return __builtin_amdgcn_wmma_f32_16x16x32_bf16(false, a, false, b, (short)0, c, false, false);
}

// Load one 16-element bf16 WMMA fragment: two contiguous 8xbf16 (16B) runs at
// p and p+16 (ISA 16-bit layout: element e -> K = e + (e>=8?8:0) + 8*half,
// caller folds the 8*half offset into p).
__device__ __forceinline__ bf16x16 frag_ld(const __bf16* p) {
    bf16x8 lo = *reinterpret_cast<const bf16x8*>(p);
    bf16x8 hi = *reinterpret_cast<const bf16x8*>(p + 16);
    bf16x16 f;
#pragma unroll
    for (int e = 0; e < 8; ++e) { f[e] = lo[e]; f[e + 8] = hi[e]; }
    return f;
}

// ---------------------------------------------------------------------------
// One-time precision/layout conversion kernels
// ---------------------------------------------------------------------------
__global__ void cvt_f32_bf16(const float* __restrict__ src, __bf16* __restrict__ dst, int n)
{
    int i = blockIdx.x * 256 + threadIdx.x;
    if (i < n) dst[i] = (__bf16)src[i];
}

// dst[n*K + k] = (bf16)src[k*Ncols + n]   (weight transpose + down-convert)
__global__ void transpose_cvt(const float* __restrict__ src, __bf16* __restrict__ dst,
                              int K, int Ncols)
{
    int i = blockIdx.x * 256 + threadIdx.x;
    if (i < K * Ncols) {
        int n = i / K, k = i - n * K;
        dst[i] = (__bf16)src[(size_t)k * Ncols + n];
    }
}

// ---------------------------------------------------------------------------
// LDS-staged 128x128 GEMM core (shared by qkv_gemm / out_gemm).
// Block: 256 threads = 8 waves (4 row-groups x 2 col-groups), wave = 32x64.
// A: [M,768] bf16 row-major; Bt: [N,768] bf16 (transposed weights).
// Accumulators acc[rt(2)][ct(4)] left in registers for the caller's epilogue.
// ---------------------------------------------------------------------------
struct GemmFrag {
    f32x8 acc[2][4];
    int mrow0;      // first row of this wave's tile
    int ncol0;      // first col of this wave's tile
};

__device__ __forceinline__ void gemm_core_128x128(
    const __bf16* __restrict__ A, const __bf16* __restrict__ Bt,
    int Mbase, int Nbase, __bf16* As, __bf16* Bs, GemmFrag& G)
{
    const int tid  = threadIdx.x;
    const int wave = tid >> 5;
    const int lane = tid & 31;
    const int half = lane >> 4;
    const int lr   = lane & 15;
    const int waveM = wave >> 1;          // 0..3  (32-row groups)
    const int waveN = wave & 1;           // 0..1  (64-col groups)
    const int rbase = waveM * 32;
    const int cbase = waveN * 64;

    G.mrow0 = Mbase + rbase;
    G.ncol0 = Nbase + cbase;
#pragma unroll
    for (int rt = 0; rt < 2; ++rt)
#pragma unroll
        for (int ct = 0; ct < 4; ++ct) G.acc[rt][ct] = (f32x8){};

    // staging assignment: 2 threads per tile row, 16 bf16 (32B) each
    const int srow = tid >> 1;            // 0..127
    const int sko  = (tid & 1) * 16;      // 0 or 16
    const __bf16* gA = A  + (size_t)(Mbase + srow) * Dv + sko;
    const __bf16* gB = Bt + (size_t)(Nbase + srow) * Dv + sko;

    bf16x8 a_lo = *reinterpret_cast<const bf16x8*>(gA);
    bf16x8 a_hi = *reinterpret_cast<const bf16x8*>(gA + 8);
    bf16x8 b_lo = *reinterpret_cast<const bf16x8*>(gB);
    bf16x8 b_hi = *reinterpret_cast<const bf16x8*>(gB + 8);

#pragma unroll 1
    for (int kt = 0; kt < Dv / 32; ++kt) {
        __syncthreads();                 // previous tile fully consumed
        *reinterpret_cast<bf16x8*>(&As[srow * LSTR + sko])     = a_lo;
        *reinterpret_cast<bf16x8*>(&As[srow * LSTR + sko + 8]) = a_hi;
        *reinterpret_cast<bf16x8*>(&Bs[srow * LSTR + sko])     = b_lo;
        *reinterpret_cast<bf16x8*>(&Bs[srow * LSTR + sko + 8]) = b_hi;
        __syncthreads();                 // tile visible to all waves

        if (kt + 1 < Dv / 32) {          // prefetch next tile (overlaps WMMAs)
            int k0 = (kt + 1) * 32;
            a_lo = *reinterpret_cast<const bf16x8*>(gA + k0);
            a_hi = *reinterpret_cast<const bf16x8*>(gA + k0 + 8);
            b_lo = *reinterpret_cast<const bf16x8*>(gB + k0);
            b_hi = *reinterpret_cast<const bf16x8*>(gB + k0 + 8);
        }

        bf16x16 af0 = frag_ld(&As[(rbase +  0 + lr) * LSTR + 8 * half]);
        bf16x16 af1 = frag_ld(&As[(rbase + 16 + lr) * LSTR + 8 * half]);
#pragma unroll
        for (int ct = 0; ct < 4; ++ct) {
            bf16x16 bfv = frag_ld(&Bs[(cbase + ct * 16 + lr) * LSTR + 8 * half]);
            G.acc[0][ct] = wmma_bf16(af0, bfv, G.acc[0][ct]);
            G.acc[1][ct] = wmma_bf16(af1, bfv, G.acc[1][ct]);
        }
    }
}

// ---------------------------------------------------------------------------
// Kernel 1: qkv = X @ W_qkv + b_qkv, scattered to padded bf16 Q/K/V^T layouts
//   Xbf: [8192,768] bf16, Wt: [2304,768] bf16 (transposed W_qkv)
//   Qbf,Kbf: [B,H,64,32] bf16 (cols 24..31 pre-zeroed)
//   Vtb:     [B,H,32,64] bf16 (rows 24..31 pre-zeroed)
// grid: (2304/128, 8192/128), block 256
// ---------------------------------------------------------------------------
__global__ void __launch_bounds__(256, 1)
qkv_gemm(const __bf16* __restrict__ Xbf, const __bf16* __restrict__ Wt,
         const float* __restrict__ bias,
         __bf16* __restrict__ Qbf, __bf16* __restrict__ Kbf,
         __bf16* __restrict__ Vtb)
{
    __shared__ __align__(16) __bf16 As[128 * LSTR];
    __shared__ __align__(16) __bf16 Bs[128 * LSTR];
    GemmFrag G;
    gemm_core_128x128(Xbf, Wt, blockIdx.y * 128, blockIdx.x * 128, As, Bs, G);

    const int lane = threadIdx.x & 31;
    const int half = lane >> 4;
    const int lr   = lane & 15;
#pragma unroll
    for (int ct = 0; ct < 4; ++ct) {
        const int ncol  = G.ncol0 + ct * 16 + lr;
        const int which = ncol / (Hv * DKv);            // 0=q 1=k 2=v
        const int rem   = ncol - which * (Hv * DKv);
        const int h = rem / DKv, d = rem - h * DKv;
        const float bv = bias[ncol];
#pragma unroll
        for (int rt = 0; rt < 2; ++rt) {
#pragma unroll
            for (int r = 0; r < 8; ++r) {
                int mrow = G.mrow0 + rt * 16 + r + 8 * half;
                int b = mrow >> 6, n = mrow & 63;
                __bf16 z = (__bf16)(G.acc[rt][ct][r] + bv);
                if (which == 0)      Qbf[((size_t)(b * Hv + h) * Nv + n) * DKP + d] = z;
                else if (which == 1) Kbf[((size_t)(b * Hv + h) * Nv + n) * DKP + d] = z;
                else                 Vtb[((size_t)(b * Hv + h) * DKP + d) * Nv + n] = z;
            }
        }
    }
}

// ---------------------------------------------------------------------------
// Kernel 3: out = Z @ W_out + b_out   (Z: [8192,768] bf16, Wt: [768,768] bf16^T)
// grid: (768/128, 8192/128), block 256
// ---------------------------------------------------------------------------
__global__ void __launch_bounds__(256, 1)
out_gemm(const __bf16* __restrict__ Z, const __bf16* __restrict__ Wt,
         const float* __restrict__ bias, float* __restrict__ out)
{
    __shared__ __align__(16) __bf16 As[128 * LSTR];
    __shared__ __align__(16) __bf16 Bs[128 * LSTR];
    GemmFrag G;
    gemm_core_128x128(Z, Wt, blockIdx.y * 128, blockIdx.x * 128, As, Bs, G);

    const int lane = threadIdx.x & 31;
    const int half = lane >> 4;
    const int lr   = lane & 15;
#pragma unroll
    for (int ct = 0; ct < 4; ++ct) {
        const int ncol = G.ncol0 + ct * 16 + lr;
        const float bv = bias[ncol];
#pragma unroll
        for (int rt = 0; rt < 2; ++rt)
#pragma unroll
            for (int r = 0; r < 8; ++r)
                out[(size_t)(G.mrow0 + rt * 16 + r + 8 * half) * Dv + ncol] =
                    G.acc[rt][ct][r] + bv;
    }
}

// ---------------------------------------------------------------------------
// Kernel 2: fused attention core, one block per (b,h), 128 threads
// ---------------------------------------------------------------------------
__global__ void __launch_bounds__(128)
attn_core(
    const __bf16* __restrict__ Qbf, const __bf16* __restrict__ Kbf,
    const __bf16* __restrict__ Vtb,
    const int* __restrict__ conn, const int* __restrict__ dist,
    const int* __restrict__ traj, const unsigned char* __restrict__ mask,
    const float* __restrict__ edge_q, const float* __restrict__ edge_k,
    const float* __restrict__ dist_q, const float* __restrict__ dist_k,
    const float* __restrict__ path_emb, const float* __restrict__ path_pos_w,
    const float* __restrict__ edge_val, const float* __restrict__ dist_val,
    const float* __restrict__ toep_r, const float* __restrict__ toep_c,
    __bf16* __restrict__ Zbf)
{
    const int bh = blockIdx.x;
    const int b = bh >> 5, h = bh & 31;
    const int wave = threadIdx.x >> 5;
    const int lane = threadIdx.x & 31;
    const int half = lane >> 4;
    const int lr   = lane & 15;

    __shared__ float S[Nv][Nv + 1];     // scores -> attention weights
    __shared__ float QE[Nv][TEv];       // q . Ek(edge)   ; reused as s_edge
    __shared__ float KE[Nv][TEv];       // k . Eq(edge)
    __shared__ float QD[Nv][TDv];       // q . Ek(dist)   ; reused as s_dist
    __shared__ float KD[Nv][TDv];       // k . Eq(dist)

    const __bf16* Qb = Qbf + (size_t)bh * Nv * DKP;
    const __bf16* Kb = Kbf + (size_t)bh * Nv * DKP;

    // ---- phase 1: raw scores q.k^T via WMMA (wave w = row tile) ----
    {
        bf16x16 af = frag_ld(Qb + (wave * 16 + lr) * DKP + 8 * half);
        for (int jt = 0; jt < 4; ++jt) {
            bf16x16 bfv = frag_ld(Kb + (jt * 16 + lr) * DKP + 8 * half);
            f32x8 c = {};
            c = wmma_bf16(af, bfv, c);
#pragma unroll
            for (int r = 0; r < 8; ++r)
                S[wave * 16 + r + 8 * half][jt * 16 + lr] = c[r];
        }
    }

    // ---- phase 2: per-type bias tables ----
    for (int idx = threadIdx.x; idx < Nv * TEv; idx += 128) {
        int i = idx >> 5, t = idx & (TEv - 1);
        const __bf16* q = Qb + i * DKP;
        const __bf16* k = Kb + i * DKP;
        const float* ek = edge_k + (size_t)(t * Hv + h) * DKv;
        const float* eq = edge_q + (size_t)(t * Hv + h) * DKv;
        float s1 = 0.f, s2 = 0.f;
        for (int d = 0; d < DKv; ++d) {
            float qd = (float)q[d], kd = (float)k[d];
            s1 += qd * ek[d]; s2 += kd * eq[d];
        }
        QE[i][t] = s1; KE[i][t] = s2;
    }
    for (int idx = threadIdx.x; idx < Nv * TDv; idx += 128) {
        int i = idx / TDv, t = idx - i * TDv;
        const __bf16* q = Qb + i * DKP;
        const __bf16* k = Kb + i * DKP;
        const float* dk = dist_k + (size_t)(t * Hv + h) * DKv;
        const float* dq = dist_q + (size_t)(t * Hv + h) * DKv;
        float s1 = 0.f, s2 = 0.f;
        for (int d = 0; d < DKv; ++d) {
            float qd = (float)q[d], kd = (float)k[d];
            s1 += qd * dk[d]; s2 += kd * dq[d];
        }
        QD[i][t] = s1; KD[i][t] = s2;
    }
    __syncthreads();

    // ---- phase 3: biases + path bias + scale + mask + softmax + toeplitz ----
    if (threadIdx.x < Nv) {
        const int i = threadIdx.x;
        const int* crow = conn + (size_t)(b * Nv + i) * Nv;
        const int* drow = dist + (size_t)(b * Nv + i) * Nv;
        const int* trow = traj + (size_t)(b * Nv + i) * Nv * (Lv * 3);
        float mx = -__builtin_inff();
        for (int j = 0; j < Nv; ++j) {
            int te = crow[j], td = drow[j];
            float s = S[i][j] + QE[i][te] + KE[j][te] + QD[i][td] + KD[j][td];
            float pb = 0.f;
            const int* tj = trow + j * (Lv * 3);
#pragma unroll
            for (int l = 0; l < Lv; ++l) {
                float pw = path_pos_w[l * Hv + h];
                pb += (path_emb[tj[l * 3 + 0] * Hv + h] +
                       path_emb[tj[l * 3 + 1] * Hv + h] +
                       path_emb[tj[l * 3 + 2] * Hv + h]) * pw;
            }
            float denom = (float)td; if (denom < 1.f) denom = 1.f;
            s += pb / denom;
            s *= 0.20412414523193154f;            // DK^-0.5
            if (mask[b * Nv + j]) s = -__builtin_inff();
            S[i][j] = s;
            if (s > mx) mx = s;
        }
        float sum = 0.f;
        for (int j = 0; j < Nv; ++j) {
            float e = __expf(S[i][j] - mx);
            S[i][j] = e; sum += e;
        }
        float inv = (sum > 0.f) ? 1.f / sum : 0.f;
        for (int j = 0; j < Nv; ++j) {
            int off = j - i;
            float tp = (off >= 0) ? toep_r[h * Nv + off] : toep_c[h * Nv - off];
            S[i][j] *= inv * tp;
        }
    }
    __syncthreads();

    // ---- phase 4: per-type attention sums (reuse QE/QD as s_edge/s_dist) ----
    for (int idx = threadIdx.x; idx < Nv * TEv; idx += 128) {
        int i = idx >> 5, t = idx & (TEv - 1);
        const int* crow = conn + (size_t)(b * Nv + i) * Nv;
        float s = 0.f;
        for (int j = 0; j < Nv; ++j) if (crow[j] == t) s += S[i][j];
        QE[i][t] = s;
    }
    for (int idx = threadIdx.x; idx < Nv * TDv; idx += 128) {
        int i = idx / TDv, t = idx - i * TDv;
        const int* drow = dist + (size_t)(b * Nv + i) * Nv;
        float s = 0.f;
        for (int j = 0; j < Nv; ++j) if (drow[j] == t) s += S[i][j];
        QD[i][t] = s;
    }
    __syncthreads();

    // ---- phase 5: z = a @ v (WMMA) + comp-value corrections, store bf16 ----
    {
        bf16x16 a0, a1;
        const int i = wave * 16 + lr;
#pragma unroll
        for (int e = 0; e < 16; ++e) {
            int j = e + (e >= 8 ? 8 : 0) + 8 * half;
            a0[e] = (__bf16)S[i][j];
            a1[e] = (__bf16)S[i][32 + j];
        }
        for (int nt = 0; nt < 2; ++nt) {
            const int d = nt * 16 + lr;
            const __bf16* vcol = Vtb + ((size_t)bh * DKP + d) * Nv;
            bf16x16 b0 = frag_ld(vcol + 8 * half);
            bf16x16 b1 = frag_ld(vcol + 32 + 8 * half);
            f32x8 c = {};
            c = wmma_bf16(a0, b0, c);
            c = wmma_bf16(a1, b1, c);
            if (d < DKv) {
#pragma unroll
                for (int r = 0; r < 8; ++r) {
                    int i2 = wave * 16 + r + 8 * half;
                    float z = c[r];
                    for (int t = 0; t < TEv; ++t)
                        z += QE[i2][t] * edge_val[(size_t)(t * Hv + h) * DKv + d];
                    for (int t = 0; t < TDv; ++t)
                        z += QD[i2][t] * dist_val[(size_t)(t * Hv + h) * DKv + d];
                    Zbf[(size_t)(b * Nv + i2) * (Hv * DKv) + h * DKv + d] = (__bf16)z;
                }
            }
        }
    }
}

// ---------------------------------------------------------------------------
extern "C" void kernel_launch(void* const* d_in, const int* in_sizes, int n_in,
                              void* d_out, int out_size, void* d_ws, size_t ws_size,
                              hipStream_t stream)
{
    const float* node_reps = (const float*)d_in[0];
    const int*   distance  = (const int*)d_in[1];
    const int*   conn      = (const int*)d_in[2];
    const int*   traj      = (const int*)d_in[3];
    const unsigned char* mask = (const unsigned char*)d_in[4];
    const float* W_qkv  = (const float*)d_in[5];
    const float* b_qkv  = (const float*)d_in[6];
    const float* W_out  = (const float*)d_in[7];
    const float* b_out  = (const float*)d_in[8];
    const float* edge_q = (const float*)d_in[9];
    const float* edge_k = (const float*)d_in[10];
    const float* dist_q = (const float*)d_in[11];
    const float* dist_k = (const float*)d_in[12];
    const float* path_emb   = (const float*)d_in[13];
    const float* path_pos_w = (const float*)d_in[14];
    const float* edge_val   = (const float*)d_in[15];
    const float* dist_val   = (const float*)d_in[16];
    const float* toep_r     = (const float*)d_in[17];
    const float* toep_c     = (const float*)d_in[18];

    const size_t szQ   = (size_t)Bv * Hv * Nv * DKP * sizeof(__bf16);   // 16 MB
    const size_t szZ   = (size_t)Bv * Nv * Dv * sizeof(__bf16);         // 12.6 MB
    const size_t szX   = (size_t)Bv * Nv * Dv * sizeof(__bf16);         // 12.6 MB
    const size_t szWq  = (size_t)Dv * QKVN * sizeof(__bf16);            // 3.5 MB
    const size_t szWo  = (size_t)Dv * Dv * sizeof(__bf16);              // 1.2 MB

    char* p = (char*)d_ws;
    __bf16* Qbf   = (__bf16*)p;            p += szQ;
    __bf16* Kbf   = (__bf16*)p;            p += szQ;
    __bf16* Vtb   = (__bf16*)p;            p += szQ;
    __bf16* Zbf   = (__bf16*)p;            p += szZ;
    __bf16* Xbf   = (__bf16*)p;            p += szX;
    __bf16* Wqkvt = (__bf16*)p;            p += szWq;
    __bf16* Woutt = (__bf16*)p;            p += szWo;

    // zero the padded DK lanes of Q/K/V^T (cols/rows 24..31 must be 0)
    hipMemsetAsync(d_ws, 0, 3 * szQ, stream);

    // one-time precision / layout conversion
    {
        int nX = Bv * Nv * Dv;
        cvt_f32_bf16<<<dim3((nX + 255) / 256), 256, 0, stream>>>(node_reps, Xbf, nX);
        int nWq = Dv * QKVN;
        transpose_cvt<<<dim3((nWq + 255) / 256), 256, 0, stream>>>(W_qkv, Wqkvt, Dv, QKVN);
        int nWo = Dv * Dv;
        transpose_cvt<<<dim3((nWo + 255) / 256), 256, 0, stream>>>(W_out, Woutt, Dv, Dv);
    }

    qkv_gemm<<<dim3(QKVN / 128, (Bv * Nv) / 128), 256, 0, stream>>>(
        Xbf, Wqkvt, b_qkv, Qbf, Kbf, Vtb);

    attn_core<<<dim3(Bv * Hv), 128, 0, stream>>>(
        Qbf, Kbf, Vtb, conn, distance, traj, mask,
        edge_q, edge_k, dist_q, dist_k, path_emb, path_pos_w,
        edge_val, dist_val, toep_r, toep_c, Zbf);

    out_gemm<<<dim3(Dv / 128, (Bv * Nv) / 128), 256, 0, stream>>>(
        Zbf, Woutt, b_out, (float*)d_out);
}
